// PrototypeLayer_32392643346454
// MI455X (gfx1250) — matrix operand order repaired and checked by reading, not compiled, not run on hardware
//
#include <hip/hip_runtime.h>
#include <hip/hip_bf16.h>
#include <stdint.h>

// Problem constants (from reference): B=256, NP=512, C=512, H=7, W=8 -> HW=56
#define B_   256
#define NP_  512
#define C_   512
#define HW_  56

typedef __attribute__((ext_vector_type(16))) __bf16 v16bf;
typedef __attribute__((ext_vector_type(8)))  float  v8f;

// CDNA5 async global->LDS direct copy (ASYNCcnt path), guarded so the file
// still compiles (with the synchronous fallback) if the toolchain lacks it.
#if defined(__HIP_DEVICE_COMPILE__) && __has_builtin(__builtin_amdgcn_global_load_async_to_lds_b128)
#define HAVE_ASYNC_LDS 1
#else
#define HAVE_ASYNC_LDS 0
#endif

#if HAVE_ASYNC_LDS
// Builtin signature (from clang diagnostic): void(int4 AS1*, int4 AS3*, Ii, Ii)
typedef int v4i_gcc __attribute__((__vector_size__(16)));
typedef __attribute__((address_space(1))) v4i_gcc g_v4i;
typedef __attribute__((address_space(3))) v4i_gcc l_v4i;
static __device__ __forceinline__ g_v4i* as_global(const void* p) {
    return (g_v4i*)(uintptr_t)p;
}
static __device__ __forceinline__ l_v4i* as_lds(const void* p) {
    // generic shared pointer: low 32 bits are the LDS byte offset
    return (l_v4i*)(uint32_t)(uintptr_t)p;
}
#endif

static __device__ __forceinline__ void async_wait0() {
#if HAVE_ASYNC_LDS
#if __has_builtin(__builtin_amdgcn_s_wait_asynccnt)
    __builtin_amdgcn_s_wait_asynccnt(0);
#else
    asm volatile("s_wait_asynccnt 0x0" ::: "memory");
#endif
#endif
}

static __device__ __forceinline__ unsigned short f32_to_bf16_rne(float f) {
    union { float f; uint32_t u; } cv; cv.f = f;
    uint32_t u = cv.u;
    u += 0x7FFFu + ((u >> 16) & 1u);     // round-to-nearest-even
    return (unsigned short)(u >> 16);
}

// ---------------------------------------------------------------------------
// Zero-init kernel (S accumulator + BN stats live contiguously in workspace)
// ---------------------------------------------------------------------------
__global__ void proto_zero_kernel(float* __restrict__ p, int n) {
    int i = blockIdx.x * 256 + threadIdx.x;
    if (i < n) p[i] = 0.0f;
}

// ---------------------------------------------------------------------------
// Pack kernel: src [rows, C, HW] f32  ->  dst [HW][rows][C] bf16 (K-contig)
// Also computes norms[row*HW + p] = ||src[row, :, p]||_2
// One wave per (row, p) pair; 8 waves per 256-thread block.
// ---------------------------------------------------------------------------
__global__ __launch_bounds__(256)
void proto_pack_kernel(const float* __restrict__ src,
                       unsigned short* __restrict__ dst,
                       float* __restrict__ norms,
                       int nrows) {
    const int wid  = threadIdx.x >> 5;
    const int lane = threadIdx.x & 31;
    const int pair = blockIdx.x * 8 + wid;
    const int row  = pair / HW_;
    const int p    = pair % HW_;
    if (row >= nrows) return;

    const float*    s = src + (size_t)row * C_ * HW_ + p;
    unsigned short* d = dst + ((size_t)p * nrows + row) * C_;

    float ssq = 0.0f;
    #pragma unroll 4
    for (int i = 0; i < C_ / 32; ++i) {
        int   c = i * 32 + lane;
        float v = s[(size_t)c * HW_];
        ssq += v * v;
        d[c] = f32_to_bf16_rne(v);
    }
    // wave32 butterfly reduction
    #pragma unroll
    for (int off = 16; off >= 1; off >>= 1) ssq += __shfl_xor(ssq, off, 32);
    if (lane == 0) norms[row * HW_ + p] = sqrtf(ssq);
}

// ---------------------------------------------------------------------------
// WMMA GEMM + cosine + BN-stat accumulation, double-buffered async staging.
// Grid (2, 4, 56): 128x128 tile of (B x NP) for one spatial position p.
// Block = 256 threads = 8 waves; wave (wm in 0..3, wn in 0..1) owns 32x64:
//   2 (M) x 4 (N) accumulators of v_wmma_f32_16x16x32_bf16.
// Dynamic LDS: 4 panels of 128x64 bf16 (stride 72) = 73728 bytes.
// ---------------------------------------------------------------------------
#define KB_   64   // K panel per LDS stage
#define STR_  72   // padded LDS row stride (elems): breaks bank conflicts
#define PANEL_ (128 * STR_)

__global__ __launch_bounds__(256)
void proto_gemm_kernel(const unsigned short* __restrict__ Xp,
                       const unsigned short* __restrict__ Pp,
                       const float* __restrict__ xnorm,
                       const float* __restrict__ pnorm,
                       float* __restrict__ S,
                       float* __restrict__ stats) {
    extern __shared__ unsigned short smem[];
    unsigned short* Asb[2] = { smem,              smem + PANEL_     };
    unsigned short* Bsb[2] = { smem + 2 * PANEL_, smem + 3 * PANEL_ };

    const int tid    = threadIdx.x;
    const int lane   = tid & 31;
    const int wid    = tid >> 5;
    const int wave_m = wid & 3;   // 4 M-waves
    const int wave_n = wid >> 2;  // 2 N-waves
    const int half   = lane >> 4; // half-wave select
    const int l16    = lane & 15;

    const int p  = blockIdx.z;
    const int bm = blockIdx.x * 128;
    const int bn = blockIdx.y * 128;

    const unsigned short* Ag = Xp + ((size_t)p * B_  + bm) * C_;
    const unsigned short* Bg = Pp + ((size_t)p * NP_ + bn) * C_;

    // Stage one 128xKB panel pair into LDS buffer `buf` for K offset k0.
    auto stage = [&](int buf, int k0) {
        unsigned short* Al = Asb[buf];
        unsigned short* Bl = Bsb[buf];
        #pragma unroll
        for (int i = 0; i < 4; ++i) {
            int chunk = tid + i * 256;
            int row   = chunk >> 3;
            int cc    = (chunk & 7) * 8;
            const unsigned short* ga = &Ag[(size_t)row * C_ + k0 + cc];
            const unsigned short* gb = &Bg[(size_t)row * C_ + k0 + cc];
#if HAVE_ASYNC_LDS
            __builtin_amdgcn_global_load_async_to_lds_b128(
                as_global(ga), as_lds(&Al[row * STR_ + cc]), 0, 0);
            __builtin_amdgcn_global_load_async_to_lds_b128(
                as_global(gb), as_lds(&Bl[row * STR_ + cc]), 0, 0);
#else
            *(uint4*)&Al[row * STR_ + cc] = *(const uint4*)ga;
            *(uint4*)&Bl[row * STR_ + cc] = *(const uint4*)gb;
#endif
            if (i == 0 && k0 + KB_ < C_) {
                // panel k0+KB is (or will be) in async flight; warm L2 for k0+2*KB
                __builtin_prefetch(ga + KB_, 0, 1);
                __builtin_prefetch(gb + KB_, 0, 1);
            }
        }
    };

    v8f acc[2][4];
    {
        v8f z = {0.f, 0.f, 0.f, 0.f, 0.f, 0.f, 0.f, 0.f};
        #pragma unroll
        for (int i = 0; i < 2; ++i)
            #pragma unroll
            for (int j = 0; j < 4; ++j) acc[i][j] = z;
    }

    stage(0, 0);
    int buf = 0;
    for (int k0 = 0; k0 < C_; k0 += KB_) {
        async_wait0();          // this wave's async LDS writes are done
        __syncthreads();        // everyone's writes visible, prior reads done
        if (k0 + KB_ < C_) stage(buf ^ 1, k0 + KB_);   // overlap with WMMA

        const unsigned short* Al = Asb[buf];
        const unsigned short* Bl = Bsb[buf];
        #pragma unroll
        for (int kk = 0; kk < KB_; kk += 32) {
            union { uint4 q[2]; v16bf v; } a[2], b[4];
            // A 16x32 bf16 frag: lane l16 = row; half selects K runs
            // {0..7,16..23} vs {8..15,24..31}: two 16B runs (ISA 7.12.2).
            #pragma unroll
            for (int tm = 0; tm < 2; ++tm) {
                int row = wave_m * 32 + tm * 16 + l16;
                int kb  = kk + half * 8;
                a[tm].q[0] = *(const uint4*)&Al[row * STR_ + kb];
                a[tm].q[1] = *(const uint4*)&Al[row * STR_ + kb + 16];
            }
            // B 32x16 bf16 frag: lane l16 = column (N); half selects K runs
            // {0..7,16..23} vs {8..15,24..31}: two 16B runs (matches A).
            #pragma unroll
            for (int tn = 0; tn < 4; ++tn) {
                int col = wave_n * 64 + tn * 16 + l16;
                int kb  = kk + half * 8;
                b[tn].q[0] = *(const uint4*)&Bl[col * STR_ + kb];
                b[tn].q[1] = *(const uint4*)&Bl[col * STR_ + kb + 16];
            }
            #pragma unroll
            for (int tm = 0; tm < 2; ++tm)
                #pragma unroll
                for (int tn = 0; tn < 4; ++tn)
                    acc[tm][tn] = __builtin_amdgcn_wmma_f32_16x16x32_bf16(
                        false, a[tm].v, false, b[tn].v,
                        (short)0, acc[tm][tn], false, false);
        }
        buf ^= 1;
    }

    // --- Epilogue: cosine sim, accumulate S[b,j] over p, and BN stats per j ---
    float pnv[4];
    #pragma unroll
    for (int tn = 0; tn < 4; ++tn) {
        int jg = bn + wave_n * 64 + tn * 16 + l16;
        pnv[tn] = pnorm[jg * HW_ + p];
    }
    float xnv[2][8];
    #pragma unroll
    for (int tm = 0; tm < 2; ++tm)
        #pragma unroll
        for (int v = 0; v < 8; ++v) {
            int mg = bm + wave_m * 32 + tm * 16 + half * 8 + v;
            xnv[tm][v] = xnorm[mg * HW_ + p];
        }

    float cs[4] = {0.f, 0.f, 0.f, 0.f};
    float cq[4] = {0.f, 0.f, 0.f, 0.f};
    #pragma unroll
    for (int tm = 0; tm < 2; ++tm) {
        #pragma unroll
        for (int v = 0; v < 8; ++v) {
            int    mg   = bm + wave_m * 32 + tm * 16 + half * 8 + v;
            float* Srow = S + (size_t)mg * NP_;
            #pragma unroll
            for (int tn = 0; tn < 4; ++tn) {
                int   jg    = bn + wave_n * 64 + tn * 16 + l16;
                float dot   = acc[tm][tn][v];
                float denom = fmaxf(xnv[tm][v] * pnv[tn], 1e-8f);
                float s     = dot / denom;
                atomicAdd(&Srow[jg], s);
                cs[tn] += s;
                cq[tn] += s * s;
            }
        }
    }
    // fold the two half-waves (same j column) and push BN stats
    #pragma unroll
    for (int tn = 0; tn < 4; ++tn) {
        cs[tn] += __shfl_xor(cs[tn], 16, 32);
        cq[tn] += __shfl_xor(cq[tn], 16, 32);
        if (half == 0) {
            int jg = bn + wave_n * 64 + tn * 16 + l16;
            atomicAdd(&stats[jg],       cs[tn]);
            atomicAdd(&stats[NP_ + jg], cq[tn]);
        }
    }
}

// ---------------------------------------------------------------------------
// Finalize: out[b,j] = (S[b,j]/56 - mean_j) * rsqrt(var_j + 1e-5)
// ---------------------------------------------------------------------------
__global__ void proto_finalize_kernel(const float* __restrict__ S,
                                      const float* __restrict__ stats,
                                      float* __restrict__ out) {
    int b = blockIdx.x;
    int j = threadIdx.x;
    const float invN = 1.0f / (float)(B_ * HW_);
    float m   = stats[j] * invN;
    float var = stats[NP_ + j] * invN - m * m;
    float sc  = rsqrtf(var + 1e-5f);
    out[(size_t)b * NP_ + j] = (S[(size_t)b * NP_ + j] * (1.0f / HW_) - m) * sc;
}

// ---------------------------------------------------------------------------
// Workspace layout (bytes, all 256B aligned):
//   Xp    bf16 [56][256][512]          @        0   (14,680,064)
//   Pp    bf16 [56][512][512]          @ 14,680,064 (29,360,128)
//   xnorm f32  [256][56]               @ 44,040,192 (    57,344)
//   pnorm f32  [512][56]               @ 44,097,536 (   114,688)
//   S     f32  [256][512]              @ 44,212,224 (   524,288)
//   stats f32  [2][512]                @ 44,736,512 (     4,096)
// ---------------------------------------------------------------------------
extern "C" void kernel_launch(void* const* d_in, const int* in_sizes, int n_in,
                              void* d_out, int out_size, void* d_ws, size_t ws_size,
                              hipStream_t stream) {
    (void)in_sizes; (void)n_in; (void)out_size; (void)ws_size;

    const float* x     = (const float*)d_in[0];
    const float* proto = (const float*)d_in[1];
    float*       out   = (float*)d_out;
    char*        ws    = (char*)d_ws;

    unsigned short* Xp    = (unsigned short*)(ws);
    unsigned short* Pp    = (unsigned short*)(ws + 14680064);
    float*          xnorm = (float*)(ws + 44040192);
    float*          pnorm = (float*)(ws + 44097536);
    float*          S     = (float*)(ws + 44212224);
    float*          stats = (float*)(ws + 44736512);

    // Tuple output: (similarity [256*512], prototype passthrough [512*512*56])
    (void)hipMemcpyAsync(out + (size_t)B_ * NP_, proto,
                         sizeof(float) * (size_t)NP_ * C_ * HW_,
                         hipMemcpyDeviceToDevice, stream);

    // Zero S + stats (contiguous in workspace)
    const int zn = B_ * NP_ + 2 * NP_;
    proto_zero_kernel<<<(zn + 255) / 256, 256, 0, stream>>>(S, zn);

    // Pack fp32 -> bf16 K-contiguous + channel norms
    proto_pack_kernel<<<(B_  * HW_) / 8, 256, 0, stream>>>(x,     Xp, xnorm, B_);
    proto_pack_kernel<<<(NP_ * HW_) / 8, 256, 0, stream>>>(proto, Pp, pnorm, NP_);

    // 56 batched GEMMs with fused cosine + BN-stat accumulation
    dim3 grid(B_ / 128, NP_ / 128, HW_);
    const size_t lds_bytes = 4u * PANEL_ * sizeof(unsigned short);  // 73728
    proto_gemm_kernel<<<grid, 256, lds_bytes, stream>>>(Xp, Pp, xnorm, pnorm,
                                                        S, stats);

    // BN apply + spatial mean -> [256, 512]
    proto_finalize_kernel<<<B_, NP_, 0, stream>>>(S, stats, out);
}
